// MultiheadAttention_20401094656005
// MI455X (gfx1250) — matrix operand order repaired and checked by reading, not compiled, not run on hardware
//
#include <hip/hip_runtime.h>
#include <hip/hip_bf16.h>

// ---------------------------------------------------------------------------
// MultiheadAttention with sparsemax on MI455X (gfx1250).
// bf16 WMMA (v_wmma_f32_16x16x32_bf16) for all four GEMM stages, f32 accum.
// 32x64 output strip per wave (8 WMMAs / k-step, 1.5 b128 loads per WMMA).
// Sparsemax via binary-search threshold (no sort) + exact tau correction.
// ---------------------------------------------------------------------------

typedef __bf16 bf16_t;
typedef __attribute__((ext_vector_type(16))) __bf16 v16bf;
typedef __attribute__((ext_vector_type(8)))  __bf16 v8bf;
typedef __attribute__((ext_vector_type(8)))  float  v8f;

#define T_DIM  2048
#define B_DIM  2
#define E_DIM  1024
#define H_DIM  16
#define HD_DIM 64
#define M_ROWS (T_DIM * B_DIM)   // 4096 token rows
#define S_DIM  T_DIM             // source length 2048
#define BH_DIM (B_DIM * H_DIM)   // 32

// ---------------- WMMA fragment loaders (layouts per ISA 7.12.2) ------------
// A 16x32 bf16: lanes 0-15 row M=lane, K=[k0..k0+7, k0+16..k0+23];
//               lanes 16-31 row M=lane-16, K=[k0+8..k0+15, k0+24..k0+31].
static __device__ inline v16bf load_a16x32(const bf16_t* __restrict__ base, int lda) {
  const int lane  = threadIdx.x & 31;
  const int row   = lane & 15;
  const int khalf = (lane >> 4) << 3;          // 0 or 8
  const bf16_t* p = base + row * lda + khalf;
  v8bf lo = *(const v8bf*)(p);                 // K = khalf .. khalf+7
  v8bf hi = *(const v8bf*)(p + 16);            // K = khalf+16 .. khalf+23
  return __builtin_shufflevector(lo, hi, 0, 1, 2, 3, 4, 5, 6, 7,
                                         8, 9, 10, 11, 12, 13, 14, 15);
}

// B 32x16 bf16 (column n K-contiguous): lanes 0-15 col N=lane, K=[k0..k0+7, k0+16..k0+23];
//                                       lanes 16-31 col N=lane-16, K=[k0+8..k0+15, k0+24..k0+31].
static __device__ inline v16bf load_b32x16(const bf16_t* __restrict__ base, int ldb) {
  const int lane  = threadIdx.x & 31;
  const int col   = lane & 15;
  const int khalf = (lane >> 4) << 3;          // 0 or 8
  const bf16_t* p = base + col * ldb + khalf;
  v8bf lo = *(const v8bf*)(p);
  v8bf hi = *(const v8bf*)(p + 16);
  return __builtin_shufflevector(lo, hi, 0, 1, 2, 3, 4, 5, 6, 7,
                                         8, 9, 10, 11, 12, 13, 14, 15);
}

static __device__ inline v8f wmma_bf16(v16bf a, v16bf b, v8f c) {
  return __builtin_amdgcn_wmma_f32_16x16x32_bf16(false, a, false, b,
                                                 (short)0, c, false, false);
}

// ---------------- GEMM: C[M,N] = A[M,K] * B[N,K]^T (+bias) * scale ----------
// One wave -> (MT x 64) output strip; MT in {16,32}.
// M%MT==0, N%64==0, K%32==0 guaranteed by problem dims.
template <bool BF16_OUT, int MT>
__global__ void gemm_wmma_kernel(const bf16_t* __restrict__ A,
                                 const bf16_t* __restrict__ B,
                                 const float* __restrict__ bias,
                                 void* __restrict__ Cout,
                                 int M, int N, int K,
                                 int lda, int ldb, int ldc, float scale) {
  constexpr int MR = MT / 16;                   // 1 or 2 row-fragments
  const int wid    = (int)((blockIdx.x * blockDim.x + threadIdx.x) >> 5);
  const int tilesN = N >> 6;
  const int nTiles = (M / MT) * tilesN;
  if (wid >= nTiles) return;                    // wave-uniform
  const int tm = (wid / tilesN) * MT;
  const int tn = (wid % tilesN) << 6;

  v8f acc[MR][4] = {};
  const bf16_t* Abase = A + (size_t)tm * lda;
  const bf16_t* Bbase = B + (size_t)tn * ldb;

  for (int k = 0; k < K; k += 32) {
    v16bf a[MR];
#pragma unroll
    for (int r = 0; r < MR; ++r)
      a[r] = load_a16x32(Abase + (size_t)(r << 4) * lda + k, lda);
#pragma unroll
    for (int j = 0; j < 4; ++j) {
      v16bf b = load_b32x16(Bbase + (size_t)(j << 4) * ldb + k, ldb);
#pragma unroll
      for (int r = 0; r < MR; ++r)
        acc[r][j] = wmma_bf16(a[r], b, acc[r][j]);
    }
  }

  // C/D layout: VGPR i holds (M=i, N=lane) for lanes 0-15, (M=i+8, N=lane-16).
  const int lane = threadIdx.x & 31;
  const int rsub = (lane >> 4) << 3;
  const int col0 = tn + (lane & 15);
#pragma unroll
  for (int r = 0; r < MR; ++r) {
    const int row0 = tm + (r << 4) + rsub;
#pragma unroll
    for (int j = 0; j < 4; ++j) {
      const int c  = col0 + (j << 4);
      const float bv = bias ? bias[c] : 0.0f;
#pragma unroll
      for (int i = 0; i < 8; ++i) {
        const float v = (acc[r][j][i] + bv) * scale;
        const size_t idx = (size_t)(row0 + i) * ldc + c;
        if (BF16_OUT) ((bf16_t*)Cout)[idx] = (bf16_t)v;
        else          ((float*)Cout)[idx]  = v;
      }
    }
  }
}

// ---------------- Sparsemax over one 2048-wide row per workgroup ------------
__global__ void sparsemax_kernel(const float* __restrict__ scores,
                                 bf16_t* __restrict__ attn_bf,
                                 float* __restrict__ aw_accum, float inv_h) {
  __shared__ float sh[256];
  const int row = blockIdx.x;
  const int tid = threadIdx.x;
  const float* z = scores + (size_t)row * S_DIM;

  float v[8];
#pragma unroll
  for (int i = 0; i < 8; ++i) v[i] = z[tid + (i << 8)];

  // row max
  float m = v[0];
#pragma unroll
  for (int i = 1; i < 8; ++i) m = fmaxf(m, v[i]);
  sh[tid] = m; __syncthreads();
  for (int s = 128; s > 0; s >>= 1) {
    if (tid < s) sh[tid] = fmaxf(sh[tid], sh[tid + s]);
    __syncthreads();
  }
  const float zmax = sh[0]; __syncthreads();

  // binary search tau in [zmax-1, zmax]: g(tau)=sum(max(z-tau,0)) monotone dec.
  float lo = zmax - 1.0f, hi = zmax;
  for (int it = 0; it < 40; ++it) {
    const float mid = 0.5f * (lo + hi);
    float s_loc = 0.0f;
#pragma unroll
    for (int i = 0; i < 8; ++i) s_loc += fmaxf(v[i] - mid, 0.0f);
    sh[tid] = s_loc; __syncthreads();
    for (int s = 128; s > 0; s >>= 1) {
      if (tid < s) sh[tid] += sh[tid + s];
      __syncthreads();
    }
    const float g = sh[0]; __syncthreads();
    if (g >= 1.0f) lo = mid; else hi = mid;
  }

  // exact tau from the bracketed support set {z > lo}
  float cnt_loc = 0.0f, sum_loc = 0.0f;
#pragma unroll
  for (int i = 0; i < 8; ++i) {
    if (v[i] > lo) { cnt_loc += 1.0f; sum_loc += v[i]; }
  }
  sh[tid] = cnt_loc; __syncthreads();
  for (int s = 128; s > 0; s >>= 1) {
    if (tid < s) sh[tid] += sh[tid + s];
    __syncthreads();
  }
  const float k_sup = sh[0]; __syncthreads();
  sh[tid] = sum_loc; __syncthreads();
  for (int s = 128; s > 0; s >>= 1) {
    if (tid < s) sh[tid] += sh[tid + s];
    __syncthreads();
  }
  const float sum_sup = sh[0]; __syncthreads();
  const float tau = (sum_sup - 1.0f) / k_sup;

  bf16_t* ab = attn_bf  + (size_t)row * S_DIM;
  float*  aw = aw_accum + (size_t)row * S_DIM;
#pragma unroll
  for (int i = 0; i < 8; ++i) {
    const int idx = tid + (i << 8);
    const float p = fmaxf(v[i] - tau, 0.0f);
    ab[idx] = (bf16_t)p;
    aw[idx] += p * inv_h;   // heads processed sequentially on the stream
  }
}

// ---------------- small utility kernels -------------------------------------
__global__ void f32_to_bf16_kernel(const float* __restrict__ in,
                                   bf16_t* __restrict__ out, int n) {
  const int i = blockIdx.x * blockDim.x + threadIdx.x;
  if (i < n) out[i] = (bf16_t)in[i];
}

__global__ void zero_f32_kernel(float* __restrict__ p, int n) {
  const int i = blockIdx.x * blockDim.x + threadIdx.x;
  if (i < n) p[i] = 0.0f;
}

// V (rows = s*B+b, cols = h*64+d)  ->  Vt[(b*H+h)*64+d, s]  (K-contig for attn@V)
__global__ void transpose_v_kernel(const bf16_t* __restrict__ Vbf,
                                   bf16_t* __restrict__ Vt) {
  const int i = blockIdx.x * blockDim.x + threadIdx.x;   // BH*HD*S = 4,194,304
  if (i >= BH_DIM * HD_DIM * S_DIM) return;
  const int s  = i & (S_DIM - 1);
  const int d  = (i >> 11) & (HD_DIM - 1);
  const int bh = i >> 17;
  const int b  = bh >> 4;
  const int h  = bh & 15;
  Vt[i] = Vbf[(size_t)(s * B_DIM + b) * E_DIM + h * HD_DIM + d];
}

// ---------------------------------------------------------------------------
extern "C" void kernel_launch(void* const* d_in, const int* in_sizes, int n_in,
                              void* d_out, int out_size, void* d_ws, size_t ws_size,
                              hipStream_t stream) {
  (void)in_sizes; (void)n_in; (void)out_size; (void)ws_size;

  const float* q_in   = (const float*)d_in[0];   // (T,B,E)
  const float* k_in   = (const float*)d_in[1];   // (S,B,E)
  const float* v_in   = (const float*)d_in[2];   // (S,B,E)
  const float* w_in   = (const float*)d_in[3];   // (3E,E)
  const float* b_in   = (const float*)d_in[4];   // (3E,)
  const float* w_out  = (const float*)d_in[5];   // (E,E)
  const float* b_out  = (const float*)d_in[6];   // (E,)

  float* out_f   = (float*)d_out;                               // (T,B,E)
  float* attn_w  = out_f + (size_t)M_ROWS * E_DIM;              // (B,T,S)

  // ---- workspace carve-out (≈96 MB) ----
  char* ws = (char*)d_ws;
  size_t off = 0;
  auto carve = [&](size_t bytes) {
    void* p = ws + off;
    off = (off + bytes + 255) & ~(size_t)255;
    return p;
  };
  const size_t tok_bf = (size_t)M_ROWS * E_DIM * sizeof(bf16_t);      // 8 MB
  bf16_t* xq_bf   = (bf16_t*)carve(tok_bf);
  bf16_t* xk_bf   = (bf16_t*)carve(tok_bf);
  bf16_t* xv_bf   = (bf16_t*)carve(tok_bf);
  bf16_t* w_in_bf = (bf16_t*)carve((size_t)3 * E_DIM * E_DIM * sizeof(bf16_t));
  bf16_t* w_o_bf  = (bf16_t*)carve((size_t)E_DIM * E_DIM * sizeof(bf16_t));
  bf16_t* Qbf     = (bf16_t*)carve(tok_bf);
  bf16_t* Kbf     = (bf16_t*)carve(tok_bf);
  bf16_t* Vbf     = (bf16_t*)carve(tok_bf);
  bf16_t* Vt      = (bf16_t*)carve((size_t)BH_DIM * HD_DIM * S_DIM * sizeof(bf16_t));
  bf16_t* ctx_bf  = (bf16_t*)carve(tok_bf);
  float*  scores  = (float*)carve((size_t)T_DIM * S_DIM * sizeof(float));   // 16 MB (per head)
  bf16_t* attn_bf = (bf16_t*)carve((size_t)T_DIM * S_DIM * sizeof(bf16_t)); // 8 MB

  auto cvt = [&](const float* src, bf16_t* dst, int n) {
    f32_to_bf16_kernel<<<(n + 255) / 256, 256, 0, stream>>>(src, dst, n);
  };
  // 32-row tiles for the big GEMMs
  auto gemm_f32 = [&](const bf16_t* A, const bf16_t* B, const float* bias, float* C,
                      int M, int N, int K, int lda, int ldb, int ldc, float scale) {
    const int tiles = (M >> 5) * (N >> 6);
    gemm_wmma_kernel<false, 32><<<(tiles * 32 + 255) / 256, 256, 0, stream>>>(
        A, B, bias, C, M, N, K, lda, ldb, ldc, scale);
  };
  auto gemm_bf16 = [&](const bf16_t* A, const bf16_t* B, const float* bias, bf16_t* C,
                       int M, int N, int K, int lda, int ldb, int ldc, float scale) {
    const int tiles = (M >> 5) * (N >> 6);
    gemm_wmma_kernel<true, 32><<<(tiles * 32 + 255) / 256, 256, 0, stream>>>(
        A, B, bias, C, M, N, K, lda, ldb, ldc, scale);
  };
  // 16-row tiles for the narrow attn@V GEMM (N=64) to keep wave parallelism up
  auto gemm_bf16_n64 = [&](const bf16_t* A, const bf16_t* B, bf16_t* C,
                           int M, int N, int K, int lda, int ldb, int ldc) {
    const int tiles = (M >> 4) * (N >> 6);
    gemm_wmma_kernel<true, 16><<<(tiles * 32 + 255) / 256, 256, 0, stream>>>(
        A, B, nullptr, C, M, N, K, lda, ldb, ldc, 1.0f);
  };

  // 1) f32 -> bf16 staging
  cvt(q_in,  xq_bf,   M_ROWS * E_DIM);
  cvt(k_in,  xk_bf,   M_ROWS * E_DIM);
  cvt(v_in,  xv_bf,   M_ROWS * E_DIM);
  cvt(w_in,  w_in_bf, 3 * E_DIM * E_DIM);
  cvt(w_out, w_o_bf,  E_DIM * E_DIM);

  // 2) projections (q scaled by 1/sqrt(hd)=0.125, folded into bf16 store)
  gemm_bf16(xq_bf, w_in_bf,                             b_in,             Qbf,
            M_ROWS, E_DIM, E_DIM, E_DIM, E_DIM, E_DIM, 0.125f);
  gemm_bf16(xk_bf, w_in_bf + (size_t)E_DIM * E_DIM,     b_in + E_DIM,     Kbf,
            M_ROWS, E_DIM, E_DIM, E_DIM, E_DIM, E_DIM, 1.0f);
  gemm_bf16(xv_bf, w_in_bf + (size_t)2 * E_DIM * E_DIM, b_in + 2 * E_DIM, Vbf,
            M_ROWS, E_DIM, E_DIM, E_DIM, E_DIM, E_DIM, 1.0f);

  // 3) V -> per-head transposed layout for attn@V B-operand
  {
    const int n = BH_DIM * HD_DIM * S_DIM;
    transpose_v_kernel<<<(n + 255) / 256, 256, 0, stream>>>(Vbf, Vt);
  }

  // 4) zero the attn_weights accumulator region of d_out
  {
    const int n = B_DIM * T_DIM * S_DIM;
    zero_f32_kernel<<<(n + 255) / 256, 256, 0, stream>>>(attn_w, n);
  }

  // 5) per-head: scores -> sparsemax -> context
  for (int bh = 0; bh < BH_DIM; ++bh) {
    const int b = bh >> 4;
    const int h = bh & 15;
    const bf16_t* Qh = Qbf + b * E_DIM + h * HD_DIM;
    const bf16_t* Kh = Kbf + b * E_DIM + h * HD_DIM;

    gemm_f32(Qh, Kh, nullptr, scores,
             T_DIM, S_DIM, HD_DIM, B_DIM * E_DIM, B_DIM * E_DIM, S_DIM, 1.0f);

    sparsemax_kernel<<<T_DIM, 256, 0, stream>>>(
        scores, attn_bf, attn_w + (size_t)b * T_DIM * S_DIM, 1.0f / H_DIM);

    gemm_bf16_n64(attn_bf, Vt + (size_t)bh * HD_DIM * S_DIM,
                  ctx_bf + b * E_DIM + h * HD_DIM,
                  T_DIM, HD_DIM, S_DIM, S_DIM, S_DIM, B_DIM * E_DIM);
  }

  // 6) output projection with bias -> d_out
  gemm_f32(ctx_bf, w_o_bf, b_out, out_f,
           M_ROWS, E_DIM, E_DIM, E_DIM, E_DIM, E_DIM, 1.0f);
}
